// AttentionHead_32452772889249
// MI455X (gfx1250) — compile-verified, matrix operand order checked by reading, not run on hardware
//
#include <hip/hip_runtime.h>
#include <stdint.h>

// ---------------------------------------------------------------------------
// Single-head causal attention for MI455X (gfx1250, wave32).
// Path: bf16 v_wmma_f32_16x16x32_bf16 for all three matmul stages,
// f32 accumulation, f32 online (flash) softmax.
// ---------------------------------------------------------------------------

typedef __attribute__((ext_vector_type(16))) __bf16 v16bf;
typedef __attribute__((ext_vector_type(2)))  __bf16 v2bf;
typedef __attribute__((ext_vector_type(8)))  float  v8f;

union BF16Tile {
    v16bf        v;
    unsigned int u[8];
    uint4        q[2];
};

// Pack two f32 -> packed bf16 pair (low = a, high = b).
#if __has_builtin(__builtin_amdgcn_cvt_pk_bf16_f32)
__device__ __forceinline__ unsigned int pack2bf(float a, float b) {
    union { v2bf v; unsigned int u; } cv;
    cv.v = __builtin_amdgcn_cvt_pk_bf16_f32(a, b);
    return cv.u;
}
#else
__device__ __forceinline__ unsigned int pack2bf(float a, float b) {
    // round-half-up then single v_perm_b32 grabs the two high halves
    unsigned int ua = __float_as_uint(a) + 0x8000u;
    unsigned int ub = __float_as_uint(b) + 0x8000u;
    return __builtin_amdgcn_perm(ub, ua, 0x07060302u);
}
#endif

__device__ __forceinline__ unsigned short f2bf(float a) {
    return (unsigned short)((__float_as_uint(a) + 0x8000u) >> 16);
}

__device__ __forceinline__ float fast_exp2(float x) {
#if __has_builtin(__builtin_amdgcn_exp2f)
    return __builtin_amdgcn_exp2f(x);
#else
    return exp2f(x);
#endif
}

// ---------------------------------------------------------------------------
// Kernel 1: W[k][n] f32 -> Wt[w][n][k] bf16  (transpose + convert)
// ---------------------------------------------------------------------------
__global__ void wt_prep_kernel(const float* __restrict__ Wq,
                               const float* __restrict__ Wk,
                               const float* __restrict__ Wv,
                               unsigned short* __restrict__ Wt) {
    int idx = blockIdx.x * blockDim.x + threadIdx.x;   // 3*128*1024 threads
    int w = idx >> 17;            // / (128*1024)
    int n = (idx >> 10) & 127;
    int k = idx & 1023;
    const float* W = (w == 0) ? Wq : ((w == 1) ? Wk : Wv);
    Wt[idx] = f2bf(W[k * 128 + n]);
}

// ---------------------------------------------------------------------------
// Kernel 2: projections. One wave handles 16 rows of x for one weight matrix.
//  Q,K -> bf16 row-major [b*T][128];  V -> bf16 transposed Vt[b][h][t].
// ---------------------------------------------------------------------------
__global__ void __launch_bounds__(128, 1)
proj_kernel(const float* __restrict__ x,
            const unsigned short* __restrict__ Wt,
            unsigned short* __restrict__ Qb,
            unsigned short* __restrict__ Kb,
            unsigned short* __restrict__ Vt) {
    const int lane = threadIdx.x & 31;
    const int wave = (blockIdx.x * blockDim.x + threadIdx.x) >> 5;  // 0..3071
    const int w    = wave % 3;
    const int rb   = wave / 3;          // 0..1023 (16-row blocks over B*T)
    const int row0 = rb * 16;
    const int half = lane >> 4;
    const int lm   = lane & 15;
    const int kb8  = half * 8;          // A/B operand k-offset for this half

    v8f acc[8];
    v8f zero = {};
#pragma unroll
    for (int i = 0; i < 8; i++) acc[i] = zero;

    const float* xr = x + (size_t)(row0 + lm) * 1024;

    for (int kc = 0; kc < 1024; kc += 32) {
        // A tile: 16 rows x 32 k of x, converted f32 -> bf16 in A layout.
        BF16Tile A;
        const float4* xp0 = (const float4*)(xr + kc + kb8);
        const float4* xp1 = (const float4*)(xr + kc + kb8 + 16);
        float4 f0 = xp0[0], f1 = xp0[1];
        float4 f2 = xp1[0], f3 = xp1[1];
        A.u[0] = pack2bf(f0.x, f0.y); A.u[1] = pack2bf(f0.z, f0.w);
        A.u[2] = pack2bf(f1.x, f1.y); A.u[3] = pack2bf(f1.z, f1.w);
        A.u[4] = pack2bf(f2.x, f2.y); A.u[5] = pack2bf(f2.z, f2.w);
        A.u[6] = pack2bf(f3.x, f3.y); A.u[7] = pack2bf(f3.z, f3.w);
#pragma unroll
        for (int nt = 0; nt < 8; nt++) {
            BF16Tile Bt;   // column n of W = row n of Wt, contiguous over k
            const unsigned short* wr =
                Wt + ((size_t)w * 128 + nt * 16 + lm) * 1024 + kc + kb8;
            Bt.q[0] = *(const uint4*)wr;
            Bt.q[1] = *(const uint4*)(wr + 16);
            acc[nt] = __builtin_amdgcn_wmma_f32_16x16x32_bf16(
                false, A.v, false, Bt.v, (short)0, acc[nt], false, false);
        }
    }

    if (w == 2) {
        // V transposed: Vt[b][h][t]; lane's 8 rows are contiguous in t.
        const int b  = row0 >> 11;
        const int t0 = row0 & 2047;
#pragma unroll
        for (int nt = 0; nt < 8; nt++) {
            const int h = nt * 16 + lm;
            uint4 st = make_uint4(pack2bf(acc[nt][0], acc[nt][1]),
                                  pack2bf(acc[nt][2], acc[nt][3]),
                                  pack2bf(acc[nt][4], acc[nt][5]),
                                  pack2bf(acc[nt][6], acc[nt][7]));
            *(uint4*)(Vt + ((size_t)b * 128 + h) * 2048 + t0 + half * 8) = st;
        }
    } else {
        unsigned short* Dst = (w == 0) ? Qb : Kb;
#pragma unroll
        for (int nt = 0; nt < 8; nt++) {
#pragma unroll
            for (int j = 0; j < 8; j++) {
                const int row = row0 + j + half * 8;
                Dst[(size_t)row * 128 + nt * 16 + lm] = f2bf(acc[nt][j]);
            }
        }
    }
}

// ---------------------------------------------------------------------------
// Kernel 3: flash attention, one wave per 16-query tile.
//   S^T = K * Q^T        (C layout: lane = query, VGPRs/halves = keys)
//   online softmax over keys (per-lane + one shfl_xor(16))
//   O^T = Vt * P^T       (P^T packs straight into the B operand layout)
// ---------------------------------------------------------------------------
__global__ void __launch_bounds__(128, 1)
attn_kernel(const unsigned short* __restrict__ Qb,
            const unsigned short* __restrict__ Kb,
            const unsigned short* __restrict__ Vt,
            float* __restrict__ out) {
    const int lane  = threadIdx.x & 31;
    const int wave  = (blockIdx.x * blockDim.x + threadIdx.x) >> 5;  // 0..1023
    const int b     = wave >> 7;
    const int qt    = wave & 127;
    const int qbase = qt * 16;
    const int half  = lane >> 4;
    const int lm    = lane & 15;
    const int kb8   = half * 8;

    // Resident B operand: Q^T (h x 16 queries), column query = Q row (bf16).
    const unsigned short* Qrow = Qb + ((size_t)b * 2048 + qbase + lm) * 128;
    BF16Tile Bq[4];
#pragma unroll
    for (int c = 0; c < 4; c++) {
        Bq[c].q[0] = *(const uint4*)(Qrow + c * 32 + kb8);
        Bq[c].q[1] = *(const uint4*)(Qrow + c * 32 + 16 + kb8);
    }

    v8f oacc[8];
    v8f zero = {};
#pragma unroll
    for (int i = 0; i < 8; i++) oacc[i] = zero;

    float m = -1e30f, l = 0.0f;
    // fold reference scale C^-0.5 = 1/32 and log2(e) so we can use exp2
    const float cscale = 0.03125f * 1.44269504088896340736f;

    const unsigned short* Kbb = Kb + (size_t)b * 2048 * 128;
    const unsigned short* Vbb = Vt + (size_t)b * 128 * 2048;
    const int nblocks = (qbase + 16 + 31) >> 5;   // 32 keys per block

    for (int ib = 0; ib < nblocks; ib++) {
        const int kb = ib * 32;
        v8f s0 = zero, s1 = zero;   // S^T tiles: keys kb..kb+15 / kb+16..kb+31
#pragma unroll
        for (int c = 0; c < 4; c++) {
            BF16Tile Ka0, Ka1;      // A = K rows (contiguous over h)
            const unsigned short* kr0 = Kbb + (size_t)(kb + lm) * 128 + c * 32 + kb8;
            const unsigned short* kr1 = Kbb + (size_t)(kb + 16 + lm) * 128 + c * 32 + kb8;
            Ka0.q[0] = *(const uint4*)kr0; Ka0.q[1] = *(const uint4*)(kr0 + 16);
            Ka1.q[0] = *(const uint4*)kr1; Ka1.q[1] = *(const uint4*)(kr1 + 16);
            s0 = __builtin_amdgcn_wmma_f32_16x16x32_bf16(
                false, Ka0.v, false, Bq[c].v, (short)0, s0, false, false);
            s1 = __builtin_amdgcn_wmma_f32_16x16x32_bf16(
                false, Ka1.v, false, Bq[c].v, (short)0, s1, false, false);
        }
        s0 = s0 * cscale;
        s1 = s1 * cscale;

        if (ib == nblocks - 1) {    // diagonal block: causal mask
            const int q = qbase + lm;
#pragma unroll
            for (int j = 0; j < 8; j++) {
                const int k0 = kb + j + half * 8;
                const int k1 = kb + 16 + j + half * 8;
                s0[j] = (k0 > q) ? -1e30f : s0[j];
                s1[j] = (k1 > q) ? -1e30f : s1[j];
            }
        }

        // per-query (per-lane) max over this lane's 16 keys, then cross-half
        float tmax = s0[0];
#pragma unroll
        for (int j = 1; j < 8; j++) tmax = fmaxf(tmax, s0[j]);
#pragma unroll
        for (int j = 0; j < 8; j++) tmax = fmaxf(tmax, s1[j]);
        tmax = fmaxf(tmax, __shfl_xor(tmax, 16));

        const float nm    = fmaxf(m, tmax);
        const float alpha = fast_exp2(m - nm);
        m = nm;

        float p0[8], p1[8], tsum = 0.0f;
#pragma unroll
        for (int j = 0; j < 8; j++) { p0[j] = fast_exp2(s0[j] - m); tsum += p0[j]; }
#pragma unroll
        for (int j = 0; j < 8; j++) { p1[j] = fast_exp2(s1[j] - m); tsum += p1[j]; }
        tsum += __shfl_xor(tsum, 16);
        l = l * alpha + tsum;

        // P^T C-layout packs directly into the 32-K bf16 B-operand layout.
        BF16Tile Bp;
        Bp.u[0] = pack2bf(p0[0], p0[1]); Bp.u[1] = pack2bf(p0[2], p0[3]);
        Bp.u[2] = pack2bf(p0[4], p0[5]); Bp.u[3] = pack2bf(p0[6], p0[7]);
        Bp.u[4] = pack2bf(p1[0], p1[1]); Bp.u[5] = pack2bf(p1[2], p1[3]);
        Bp.u[6] = pack2bf(p1[4], p1[5]); Bp.u[7] = pack2bf(p1[6], p1[7]);

#pragma unroll
        for (int mt = 0; mt < 8; mt++) {
            BF16Tile Va;           // A = Vt rows (h), contiguous over keys
            const unsigned short* vr =
                Vbb + (size_t)(mt * 16 + lm) * 2048 + kb + kb8;
            Va.q[0] = *(const uint4*)vr;
            Va.q[1] = *(const uint4*)(vr + 16);
            oacc[mt] = oacc[mt] * alpha;      // per-lane rescale (alpha = per-query)
            oacc[mt] = __builtin_amdgcn_wmma_f32_16x16x32_bf16(
                false, Va.v, false, Bp.v, (short)0, oacc[mt], false, false);
        }
    }

    const float rl = 1.0f / l;
    float* orow = out + ((size_t)b * 2048 + qbase + lm) * 128;
#pragma unroll
    for (int mt = 0; mt < 8; mt++) {
        float4 lo = make_float4(oacc[mt][0] * rl, oacc[mt][1] * rl,
                                oacc[mt][2] * rl, oacc[mt][3] * rl);
        float4 hi = make_float4(oacc[mt][4] * rl, oacc[mt][5] * rl,
                                oacc[mt][6] * rl, oacc[mt][7] * rl);
        *(float4*)(orow + mt * 16 + half * 8 + 0) = lo;
        *(float4*)(orow + mt * 16 + half * 8 + 4) = hi;
    }
}

// ---------------------------------------------------------------------------
// Launch: prep weights -> projections -> flash attention (all on `stream`).
// Workspace layout (bf16 = unsigned short):
//   Wt : 3*128*1024            (0.75 MB)
//   Qb : 8*2048*128            (4 MB)
//   Kb : 8*2048*128            (4 MB)
//   Vt : 8*128*2048            (4 MB)
// ---------------------------------------------------------------------------
extern "C" void kernel_launch(void* const* d_in, const int* in_sizes, int n_in,
                              void* d_out, int out_size, void* d_ws, size_t ws_size,
                              hipStream_t stream) {
    const float* x  = (const float*)d_in[0];
    const float* Wq = (const float*)d_in[1];
    const float* Wk = (const float*)d_in[2];
    const float* Wv = (const float*)d_in[3];
    float* out = (float*)d_out;

    unsigned short* Wt = (unsigned short*)d_ws;
    unsigned short* Qb = Wt + (size_t)3 * 128 * 1024;
    unsigned short* Kb = Qb + (size_t)8 * 2048 * 128;
    unsigned short* Vt = Kb + (size_t)8 * 2048 * 128;

    // 3*128*1024 = 393216 elements
    wt_prep_kernel<<<1536, 256, 0, stream>>>(Wq, Wk, Wv, Wt);
    // 1024 row-blocks * 3 weights = 3072 waves, 4 waves/block
    proj_kernel<<<768, 128, 0, stream>>>(x, Wt, Qb, Kb, Vt);
    // 8 batches * 128 query tiles = 1024 waves, 4 waves/block
    attn_kernel<<<256, 128, 0, stream>>>(Qb, Kb, Vt, out);
}
